// CoupledGraphODE_31980326486311
// MI455X (gfx1250) — matrix operand order repaired and checked by reading, not compiled, not run on hardware
//
#include <hip/hip_runtime.h>

// MI455X / gfx1250, wave32. WMMA f16->f32 for all matmuls.
// Fragment layouts (ISA 7.12.2, 16-bit, wave32):
//  A (16x32, MxK): lane<16:  j=0..7 -> K=0..7,  j=8..15 -> K=16..23
//                  lane>=16: j=0..7 -> K=8..15, j=8..15 -> K=24..31
//    -> element runs are 8 consecutive K values -> two 16B loads.
//  B (32x16, KxN): col N=l15; lane<16 holds K=0..7,16..23; lane>=16 holds K=8..15,24..31.
//  C/D (16x16 f32): elem r -> (M = r + lh*8, N = l15).

typedef __attribute__((ext_vector_type(16))) _Float16 v16h;
typedef __attribute__((ext_vector_type(8)))  _Float16 v8h;
typedef __attribute__((ext_vector_type(4)))  _Float16 v4h;
typedef __attribute__((ext_vector_type(8)))  float    v8f;

#define HD 64   // hidden dim H
#define DD 32   // input/output dim D

static __device__ __forceinline__ float sigmoidf_(float z) { return 1.0f / (1.0f + __expf(-z)); }
static __device__ __forceinline__ float siluf_(float z)    { return z * sigmoidf_(z); }

// load 8 consecutive f32 (16B-aligned) and convert to f16
static __device__ __forceinline__ v8h load8_cvt(const float* __restrict__ p)
{
    float4 u = *(const float4*)p;
    float4 v = *(const float4*)(p + 4);
    v8h d;
    d[0] = (_Float16)u.x; d[1] = (_Float16)u.y; d[2] = (_Float16)u.z; d[3] = (_Float16)u.w;
    d[4] = (_Float16)v.x; d[5] = (_Float16)v.y; d[6] = (_Float16)v.z; d[7] = (_Float16)v.w;
    return d;
}

static __device__ __forceinline__ v16h cat8(v8h lo, v8h hi)
{
    return __builtin_shufflevector(lo, hi, 0,1,2,3,4,5,6,7,8,9,10,11,12,13,14,15);
}

// B fragment from LDS pre-swizzled storage: 2x ds_load_b128
static __device__ __forceinline__ v16h ldsB(const _Float16* __restrict__ base, int tile, int lane)
{
    const v8h* p = (const v8h*)(base + ((size_t)tile * 32 + lane) * 16);
    return cat8(p[0], p[1]);
}

// ---------------------------------------------------------------------------
// Generic WMMA GEMM: C[M, NT*16] = A[M,K] @ B[K, NT*16] (+bias), K in {32,64}.
// One wave computes a 16 x (NT*16) strip. B staged in LDS in fragment order.
// ---------------------------------------------------------------------------
template<int NT>
__global__ void __launch_bounds__(256)
gemm_wmma_kernel(const float* __restrict__ A, const float* __restrict__ B,
                 const float* __restrict__ bias, float* __restrict__ C,
                 int M, int K)
{
    const int N = NT * 16;
    __shared__ _Float16 sB[64 * 64];   // fragment order: ((kb*NT+nt)*32+lane)*16+j
    for (int idx = threadIdx.x; idx < K * N; idx += blockDim.x) {
        int k = idx / N, n = idx - k * N;
        int kb = k >> 5, krem = k & 31;
        int lh = (krem >> 3) & 1, j = (krem & 7) + ((krem >> 4) << 3);
        int nt = n >> 4, l = n & 15;
        sB[(((size_t)kb * NT + nt) * 32 + lh * 16 + l) * 16 + j] = (_Float16)B[idx];
    }
    __syncthreads();

    int wave = (int)((blockIdx.x * (unsigned)blockDim.x + threadIdx.x) >> 5);
    int lane = threadIdx.x & 31;
    int l15 = lane & 15, lh = lane >> 4;
    int row0 = wave * 16;
    if (row0 >= M) return;

    v8f acc[NT];
#pragma unroll
    for (int nt = 0; nt < NT; ++nt) acc[nt] = (v8f){0,0,0,0,0,0,0,0};

    for (int kb = 0; kb * 32 < K; ++kb) {
        const float* ar = A + (size_t)(row0 + l15) * K + kb * 32 + lh * 8;
        v16h a = cat8(load8_cvt(ar), load8_cvt(ar + 16));
#pragma unroll
        for (int nt = 0; nt < NT; ++nt) {
            v16h b = ldsB(sB, kb * NT + nt, lane);
            acc[nt] = __builtin_amdgcn_wmma_f32_16x16x32_f16(
                false, a, false, b, (short)0, acc[nt], false, false);
        }
    }
#pragma unroll
    for (int nt = 0; nt < NT; ++nt) {
        int col = nt * 16 + l15;
        float bv = bias ? bias[col] : 0.0f;
#pragma unroll
        for (int r = 0; r < 8; ++r)
            C[(size_t)(row0 + lh * 8 + r) * N + col] = acc[nt][r] + bv;
    }
}

// ---------------------------------------------------------------------------
// Fused edge encoder MLP, f16 node-state mirror as input:
//   ev[e] = sigmoid( silu( silu([xh[src]|xh[dst]] @ W1 + b1) @ W2a + b2a ) @ W2b + b2b )
// One wave = 16 edges. Gather fused into stage-1 A-fragment load: each
// 8-element run is one global_load_b128 of f16 (no converts, half the bytes).
// W1/W2a in LDS in fragment order; stage-1 activations restaged per-wave in
// row-major f16 so stage-2 A fragments are 2x ds_load_b128.
// ---------------------------------------------------------------------------
__global__ void __launch_bounds__(256)
edge_mlp_kernel(const _Float16* __restrict__ xh,
                const int* __restrict__ src, const int* __restrict__ dst,
                const float* __restrict__ W1,  const float* __restrict__ b1,
                const float* __restrict__ W2a, const float* __restrict__ b2a,
                const float* __restrict__ W2b, const float* __restrict__ b2b,
                float* __restrict__ ev, int E)
{
    __shared__ _Float16 sW1[128 * 64];   // 16 tiles (kb 0..3, nt 0..3), fragment order
    __shared__ _Float16 sW2a[64 * 32];   // 4 tiles (kb 0..1, nt 0..1), fragment order
    __shared__ _Float16 sE1[8 * 16 * 64]; // per-wave stage-1 activations [row][K]

    for (int idx = threadIdx.x; idx < 128 * 64; idx += blockDim.x) {
        int k = idx >> 6, n = idx & 63;
        int kb = k >> 5, krem = k & 31;
        int lh = (krem >> 3) & 1, j = (krem & 7) + ((krem >> 4) << 3);
        int nt = n >> 4, l = n & 15;
        sW1[(((size_t)kb * 4 + nt) * 32 + lh * 16 + l) * 16 + j] = (_Float16)W1[idx];
    }
    for (int idx = threadIdx.x; idx < 64 * 32; idx += blockDim.x) {
        int k = idx >> 5, n = idx & 31;
        int kb = k >> 5, krem = k & 31;
        int lh = (krem >> 3) & 1, j = (krem & 7) + ((krem >> 4) << 3);
        int nt = n >> 4, l = n & 15;
        sW2a[(((size_t)kb * 2 + nt) * 32 + lh * 16 + l) * 16 + j] = (_Float16)W2a[idx];
    }
    __syncthreads();

    int wib  = threadIdx.x >> 5;
    int lane = threadIdx.x & 31;
    int l15 = lane & 15, lh = lane >> 4;
    int tile = blockIdx.x * 8 + wib;
    if (tile * 16 >= E) return;

    int e  = tile * 16 + l15;
    int si = src[e], di = dst[e];
    const _Float16* xs_ = xh + (size_t)si * HD;
    const _Float16* xd_ = xh + (size_t)di * HD;

    // ---- stage 1: [16,128] @ [128,64] ----
    v8f a1[4];
#pragma unroll
    for (int nt = 0; nt < 4; ++nt) a1[nt] = (v8f){0,0,0,0,0,0,0,0};
#pragma unroll
    for (int kb = 0; kb < 4; ++kb) {
        // element runs: K = kb*32 + lh*8 + {0..7} and kb*32 + 16 + lh*8 + {0..7}
        int c0 = kb * 32 + lh * 8;
        int c1 = c0 + 16;
        const _Float16* p0 = (c0 < HD) ? (xs_ + c0) : (xd_ + (c0 - HD));
        const _Float16* p1 = (c1 < HD) ? (xs_ + c1) : (xd_ + (c1 - HD));
        v16h a = cat8(*(const v8h*)p0, *(const v8h*)p1);   // 2x global_load_b128
#pragma unroll
        for (int nt = 0; nt < 4; ++nt) {
            v16h b = ldsB(sW1, kb * 4 + nt, lane);
            a1[nt] = __builtin_amdgcn_wmma_f32_16x16x32_f16(
                false, a, false, b, (short)0, a1[nt], false, false);
        }
    }
    // silu(+b1), restage to per-wave LDS (row-major f16 [16][64])
    _Float16* myE1 = sE1 + (size_t)wib * 16 * 64;
#pragma unroll
    for (int nt = 0; nt < 4; ++nt) {
        int col = nt * 16 + l15;
        float bb = b1[col];
#pragma unroll
        for (int r = 0; r < 8; ++r)
            myE1[(lh * 8 + r) * 64 + col] = (_Float16)siluf_(a1[nt][r] + bb);
    }

    // ---- stage 2: [16,64] @ [64,32]; A fragments contiguous in LDS ----
    v8f a2[2];
    a2[0] = (v8f){0,0,0,0,0,0,0,0};
    a2[1] = (v8f){0,0,0,0,0,0,0,0};
#pragma unroll
    for (int kb = 0; kb < 2; ++kb) {
        const v8h* ap = (const v8h*)(myE1 + l15 * 64 + kb * 32 + lh * 8);
        v16h a = cat8(ap[0], ap[2]);   // +0 and +16 halves
#pragma unroll
        for (int nt = 0; nt < 2; ++nt) {
            v16h b = ldsB(sW2a, kb * 2 + nt, lane);
            a2[nt] = __builtin_amdgcn_wmma_f32_16x16x32_f16(
                false, a, false, b, (short)0, a2[nt], false, false);
        }
    }

    // ---- stage 3: silu, dot with W2b[32], +b2b, sigmoid ----
    float wlo = W2b[l15], whi = W2b[16 + l15];
    float blo = b2a[l15], bhi = b2a[16 + l15];
    float bb2 = b2b[0];
#pragma unroll
    for (int r = 0; r < 8; ++r) {
        float p = siluf_(a2[0][r] + blo) * wlo + siluf_(a2[1][r] + bhi) * whi;
        p += __shfl_xor(p, 1, 32);
        p += __shfl_xor(p, 2, 32);
        p += __shfl_xor(p, 4, 32);
        p += __shfl_xor(p, 8, 32);
        if (l15 == 0)
            ev[tile * 16 + lh * 8 + r] = sigmoidf_(p + bb2);
    }
}

// ---------------------------------------------------------------------------
// scalar helper kernels
// ---------------------------------------------------------------------------
__global__ void zero_kernel(float* __restrict__ p, int n)
{
    int i = blockIdx.x * blockDim.x + threadIdx.x;
    if (i < n) p[i] = 0.0f;
}

__global__ void saxpy_kernel(float* __restrict__ out, const float* __restrict__ a,
                             const float* __restrict__ b, float s, int n)
{
    int i = blockIdx.x * blockDim.x + threadIdx.x;
    if (i < n) out[i] = a[i] + s * b[i];
}

// RK4 stage bookkeeping: acc = first ? k : acc + sacc*k;  xs = x + sxs*k
__global__ void rk4_mix_kernel(float* __restrict__ acc, float* __restrict__ xs,
                               const float* __restrict__ x, const float* __restrict__ k,
                               float sacc, float sxs, int first, int n)
{
    int i = blockIdx.x * blockDim.x + threadIdx.x;
    if (i >= n) return;
    float kv = k[i];
    acc[i] = first ? kv : acc[i] + sacc * kv;
    xs[i]  = x[i] + sxs * kv;
}

// f32 -> f16 mirror (vectorized x4)
__global__ void cvt_f16_kernel(const float* __restrict__ x, _Float16* __restrict__ o, int n4)
{
    int i = blockIdx.x * blockDim.x + threadIdx.x;
    if (i >= n4) return;
    float4 u = ((const float4*)x)[i];
    v4h d;
    d[0] = (_Float16)u.x; d[1] = (_Float16)u.y; d[2] = (_Float16)u.z; d[3] = (_Float16)u.w;
    ((v4h*)o)[i] = d;
}

__global__ void silu_kernel(const float* __restrict__ x, float* __restrict__ o, int n)
{
    int i = blockIdx.x * blockDim.x + threadIdx.x;
    if (i < n) o[i] = siluf_(x[i]);
}

__global__ void deg_kernel(const int* __restrict__ dst, const float* __restrict__ ew,
                           float* __restrict__ deg, int E)
{
    int e = blockIdx.x * blockDim.x + threadIdx.x;
    if (e < E) atomicAdd(&deg[dst[e]], ew ? ew[e] : 1.0f);
}

__global__ void dinv_kernel(float* __restrict__ deg, int N)
{
    int i = blockIdx.x * blockDim.x + threadIdx.x;
    if (i < N) { float d = deg[i] + 1.0f; deg[i] = (d > 0.0f) ? rsqrtf(d) : 0.0f; }
}

// agg[dst] += dinv[src]*ew*dinv[dst] * h[src]; one thread per (edge, channel)
__global__ void scatter_agg_kernel(const float* __restrict__ h,
                                   const int* __restrict__ src, const int* __restrict__ dst,
                                   const float* __restrict__ ew, const float* __restrict__ dinv,
                                   float* __restrict__ agg, int E, int logC)
{
    long t = (long)blockIdx.x * blockDim.x + threadIdx.x;
    long total = (long)E << logC;
    if (t >= total) return;
    int e = (int)(t >> logC);
    int c = (int)(t & ((1 << logC) - 1));
    int si = src[e], di = dst[e];
    float w = dinv[si] * dinv[di] * (ew ? ew[e] : 1.0f);
    atomicAdd(&agg[((size_t)di << logC) + c], w * h[((size_t)si << logC) + c]);
}

// out = agg + dinv^2*h + bias   (C = 1<<logC channels)
__global__ void finish_gcn_kernel(const float* __restrict__ agg, const float* __restrict__ h,
                                  const float* __restrict__ dinv, const float* __restrict__ bias,
                                  float* __restrict__ out, int N, int logC)
{
    long t = (long)blockIdx.x * blockDim.x + threadIdx.x;
    long total = (long)N << logC;
    if (t >= total) return;
    int i = (int)(t >> logC);
    int c = (int)(t & ((1 << logC) - 1));
    float d2 = dinv[i] * dinv[i];
    out[t] = agg[t] + d2 * h[t] + (bias ? bias[c] : 0.0f);
}

// LayerNorm over H=64, no affine; one wave per row.
__global__ void layernorm_kernel(const float* __restrict__ x, float* __restrict__ xn, int N)
{
    int wave = (int)((blockIdx.x * (unsigned)blockDim.x + threadIdx.x) >> 5);
    int lane = threadIdx.x & 31;
    if (wave >= N) return;
    const float* r = x + (size_t)wave * HD;
    float a = r[lane], b = r[lane + 32];
    float s = a + b, q = a * a + b * b;
#pragma unroll
    for (int m = 1; m < 32; m <<= 1) { s += __shfl_xor(s, m, 32); q += __shfl_xor(q, m, 32); }
    float mu = s * (1.0f / HD);
    float var = q * (1.0f / HD) - mu * mu;
    float rs = rsqrtf(var + 1e-5f);
    float* o = xn + (size_t)wave * HD;
    o[lane] = (a - mu) * rs;
    o[lane + 32] = (b - mu) * rs;
}

// row-wise L2 normalize in place; one wave per row.
__global__ void l2norm_kernel(float* __restrict__ x, int N)
{
    int wave = (int)((blockIdx.x * (unsigned)blockDim.x + threadIdx.x) >> 5);
    int lane = threadIdx.x & 31;
    if (wave >= N) return;
    float* r = x + (size_t)wave * HD;
    float a = r[lane], b = r[lane + 32];
    float q = a * a + b * b;
#pragma unroll
    for (int m = 1; m < 32; m <<= 1) q += __shfl_xor(q, m, 32);
    float inv = 1.0f / fmaxf(sqrtf(q), 1e-12f);
    r[lane] = a * inv;
    r[lane + 32] = b * inv;
}

// k = sigmoid(alpha_p)*0.5*((agg + dinv^2*h + b_node) - xn) - 2*xs + g2
__global__ void ode_combine_kernel(const float* __restrict__ agg, const float* __restrict__ h,
                                   const float* __restrict__ dinv, const float* __restrict__ b_node,
                                   const float* __restrict__ xn, const float* __restrict__ xs,
                                   const float* __restrict__ g2, const float* __restrict__ alpha_p,
                                   float* __restrict__ k, int N)
{
    long t = (long)blockIdx.x * blockDim.x + threadIdx.x;
    long total = (long)N * HD;
    if (t >= total) return;
    int i = (int)(t >> 6);
    int c = (int)(t & 63);
    float d2 = dinv[i] * dinv[i];
    float gn = agg[t] + d2 * h[t] + b_node[c] - xn[t];
    float alpha = sigmoidf_(alpha_p[i]);
    k[t] = alpha * 0.5f * gn - 2.0f * xs[t] + g2[t];
}

// w = (w_cpl * clip(d_p,0,1)) @ w_cpl^T   (64x64)
__global__ void build_w_kernel(const float* __restrict__ wc, const float* __restrict__ dp,
                               float* __restrict__ w)
{
    int idx = blockIdx.x * blockDim.x + threadIdx.x;
    if (idx >= 64 * 64) return;
    int i = idx >> 6, j = idx & 63;
    float s = 0.0f;
    for (int k = 0; k < 64; ++k) {
        float d = fminf(fmaxf(dp[k], 0.0f), 1.0f);
        s += wc[i * 64 + k] * d * wc[j * 64 + k];
    }
    w[idx] = s;
}

// ---------------------------------------------------------------------------
// host orchestration
// ---------------------------------------------------------------------------
static inline int nblk(long n, int t) { return (int)((n + t - 1) / t); }

extern "C" void kernel_launch(void* const* d_in, const int* in_sizes, int n_in,
                              void* d_out, int out_size, void* d_ws, size_t ws_size,
                              hipStream_t stream)
{
    const float* emb    = (const float*)d_in[0];
    const int*   eidx   = (const int*)d_in[1];
    const float* W_in   = (const float*)d_in[2];
    const float* b_in   = (const float*)d_in[3];
    const float* W_out  = (const float*)d_in[4];
    const float* b_out  = (const float*)d_in[5];
    const float* W_node = (const float*)d_in[6];
    const float* b_node = (const float*)d_in[7];
    const float* W1     = (const float*)d_in[8];
    const float* b1     = (const float*)d_in[9];
    const float* W2a    = (const float*)d_in[10];
    const float* b2a    = (const float*)d_in[11];
    const float* W2b    = (const float*)d_in[12];
    const float* b2b    = (const float*)d_in[13];
    const float* alphap = (const float*)d_in[14];
    const float* w_cpl  = (const float*)d_in[15];
    const float* d_p    = (const float*)d_in[16];

    const int N = in_sizes[0] / DD;      // 20000
    const int E = in_sizes[1] / 2;       // 320000
    const int NH = N * HD;
    const int* src = eidx;
    const int* dst = eidx + E;

    float* ws    = (float*)d_ws;
    float* xb    = ws;            // [N,H] ODE state
    float* xsb   = xb   + NH;     // stage input
    float* kbf   = xsb  + NH;     // stage derivative
    float* accb  = kbf  + NH;     // RK4 accumulator
    float* xnb   = accb + NH;     // LayerNorm output
    float* hb    = xnb  + NH;     // GEMM output
    float* aggb  = hb   + NH;     // scatter target
    float* g2b   = aggb + NH;     // x @ w
    float* sb    = g2b  + NH;     // silu(x)
    float* evb   = sb   + NH;     // [E] edge weights
    float* degb  = evb  + E;      // [N] dinv (per f-eval)
    float* dinv0 = degb + N;      // [N] dinv for ones-weights
    float* wmat  = dinv0 + N;     // [64*64]
    _Float16* xhb = (_Float16*)(wmat + 64 * 64);  // [N,H] f16 mirror of stage input
    float* outF  = (float*)d_out; // [2,N,D] ++ [N,D]

    const int T256 = 256;
    const float dt = 0.5f;        // 1/(T-1), T=3
    const int gemmBlkH = nblk((long)((N + 15) / 16) * 32, T256);

    auto run_f = [&](const float* xin, float* kout) {
        cvt_f16_kernel<<<nblk(NH / 4, T256), T256, 0, stream>>>(xin, xhb, NH / 4);
        edge_mlp_kernel<<<nblk((long)E / 16, 8), T256, 0, stream>>>(
            xhb, src, dst, W1, b1, W2a, b2a, W2b, b2b, evb, E);
        layernorm_kernel<<<nblk((long)N * 32, T256), T256, 0, stream>>>(xin, xnb, N);
        gemm_wmma_kernel<4><<<gemmBlkH, T256, 0, stream>>>(xnb, W_node, nullptr, hb, N, HD);
        zero_kernel<<<nblk(N, T256), T256, 0, stream>>>(degb, N);
        deg_kernel<<<nblk(E, T256), T256, 0, stream>>>(dst, evb, degb, E);
        dinv_kernel<<<nblk(N, T256), T256, 0, stream>>>(degb, N);
        zero_kernel<<<nblk(NH, T256), T256, 0, stream>>>(aggb, NH);
        scatter_agg_kernel<<<nblk((long)E * HD, T256), T256, 0, stream>>>(
            hb, src, dst, evb, degb, aggb, E, 6);
        gemm_wmma_kernel<4><<<gemmBlkH, T256, 0, stream>>>(xin, wmat, nullptr, g2b, N, HD);
        ode_combine_kernel<<<nblk(NH, T256), T256, 0, stream>>>(
            aggb, hb, degb, b_node, xnb, xin, g2b, alphap, kout, N);
    };

    // ---- precompute: coupling matrix + ones-degree dinv ----
    build_w_kernel<<<16, T256, 0, stream>>>(w_cpl, d_p, wmat);
    zero_kernel<<<nblk(N, T256), T256, 0, stream>>>(dinv0, N);
    deg_kernel<<<nblk(E, T256), T256, 0, stream>>>(dst, nullptr, dinv0, E);
    dinv_kernel<<<nblk(N, T256), T256, 0, stream>>>(dinv0, N);

    // ---- input GCNConv: x = gcn(emb, W_in, b_in, ones), then L2 normalize ----
    gemm_wmma_kernel<4><<<gemmBlkH, T256, 0, stream>>>(emb, W_in, nullptr, hb, N, DD);
    zero_kernel<<<nblk(NH, T256), T256, 0, stream>>>(aggb, NH);
    scatter_agg_kernel<<<nblk((long)E * HD, T256), T256, 0, stream>>>(
        hb, src, dst, nullptr, dinv0, aggb, E, 6);
    finish_gcn_kernel<<<nblk(NH, T256), T256, 0, stream>>>(aggb, hb, dinv0, b_in, xb, N, 6);
    l2norm_kernel<<<nblk((long)N * 32, T256), T256, 0, stream>>>(xb, N);

    // ---- 2 RK4 steps; output conv after each ----
    for (int step = 0; step < 2; ++step) {
        run_f(xb, kbf);                                                      // k1
        rk4_mix_kernel<<<nblk(NH, T256), T256, 0, stream>>>(
            accb, xsb, xb, kbf, 0.0f, 0.5f * dt, 1, NH);
        run_f(xsb, kbf);                                                     // k2
        rk4_mix_kernel<<<nblk(NH, T256), T256, 0, stream>>>(
            accb, xsb, xb, kbf, 2.0f, 0.5f * dt, 0, NH);
        run_f(xsb, kbf);                                                     // k3
        rk4_mix_kernel<<<nblk(NH, T256), T256, 0, stream>>>(
            accb, xsb, xb, kbf, 2.0f, dt, 0, NH);
        run_f(xsb, kbf);                                                     // k4
        saxpy_kernel<<<nblk(NH, T256), T256, 0, stream>>>(accb, accb, kbf, 1.0f, NH);
        saxpy_kernel<<<nblk(NH, T256), T256, 0, stream>>>(xb, xb, accb, dt / 6.0f, NH);

        // output conv: gcn(silu(x), W_out, b_out, ones) -> d_out[step]
        silu_kernel<<<nblk(NH, T256), T256, 0, stream>>>(xb, sb, NH);
        gemm_wmma_kernel<2><<<gemmBlkH, T256, 0, stream>>>(sb, W_out, nullptr, hb, N, HD);
        zero_kernel<<<nblk((long)N * DD, T256), T256, 0, stream>>>(aggb, N * DD);
        scatter_agg_kernel<<<nblk((long)E * DD, T256), T256, 0, stream>>>(
            hb, src, dst, nullptr, dinv0, aggb, E, 5);
        finish_gcn_kernel<<<nblk((long)N * DD, T256), T256, 0, stream>>>(
            aggb, hb, dinv0, b_out, outF + (size_t)step * N * DD, N, 5);
    }

    // out[-1] == out[1]
    hipMemcpyAsync(outF + (size_t)2 * N * DD, outF + (size_t)1 * N * DD,
                   sizeof(float) * (size_t)N * DD, hipMemcpyDeviceToDevice, stream);
}
